// F1HeteroGNN_1099511628111
// MI455X (gfx1250) — compile-verified, matrix operand order checked by reading, not run on hardware
//
#include <hip/hip_runtime.h>
#include <hip/hip_bf16.h>

// ---------------------------------------------------------------------------
// Types for CDNA5 WMMA
// ---------------------------------------------------------------------------
typedef __attribute__((ext_vector_type(16))) _Float16 v16h;
typedef __attribute__((ext_vector_type(8)))  _Float16 v8h;
typedef __attribute__((ext_vector_type(4)))  _Float16 v4h;
typedef __attribute__((ext_vector_type(8)))  float    v8f;

#define HID   256
#define BM    128
#define BN    128
#define BK    32
#define LDS_STRIDE 40   // halves; 80B/row -> 4-half offsets 8B aligned, 8-half offsets 16B aligned
#define LN_EPS 1e-5f

// ---------------------------------------------------------------------------
// Fragment load per CDNA5 16-bit A/B layout: lanes 0-15 hold K {0..7,16..23},
// lanes 16-31 hold K {8..15,24..31} of their row/col -> two contiguous 16B LDS loads.
// ---------------------------------------------------------------------------
__device__ __forceinline__ v16h frag_from(const _Float16* rowp, int half) {
  v8h lo = *(const v8h*)(rowp + half);
  v8h hi = *(const v8h*)(rowp + half + 16);
  return __builtin_shufflevector(lo, hi, 0,1,2,3,4,5,6,7,8,9,10,11,12,13,14,15);
}

__device__ __forceinline__ int imin(int a, int b) { return a < b ? a : b; }

// ---------------------------------------------------------------------------
// Generic GEMM: C[M,N] = op( A[M,K](row-major,lda) @ B[K,N](row-major,ldb) + bias )
// f32 in/out, f16 WMMA inner product with f32 accumulation.
// Requires K % 4 == 0 and N % 4 == 0 (true for all call sites: K in {16,32,256}).
// Block: 256 threads = 8 waves (4x2); block tile 128x128; wave tile 32x64 (8 wmma).
// Staging is branch-free (clamped addresses + zero-select) and software-pipelined:
// next tile's global loads are issued before the current tile's WMMA work.
// ---------------------------------------------------------------------------
__global__ __launch_bounds__(256) void gemm_wmma_f16(
    const float* __restrict__ A, int lda,
    const float* __restrict__ B, int ldb,
    const float* __restrict__ bias,
    float* __restrict__ C, int ldc,
    int M, int N, int K, int do_relu, int do_acc)
{
  __shared__ _Float16 sA[BM][LDS_STRIDE];   // A tile, row-major
  __shared__ _Float16 sBt[BN][LDS_STRIDE];  // B tile, transposed (N-major in LDS)

  const int tid  = threadIdx.x;
  const int lane = tid & 31;
  const int wid  = tid >> 5;
  const int wm   = wid & 3;       // 4 waves along M (32 rows each)
  const int wn   = wid >> 2;      // 2 waves along N (64 cols each)
  const int bm   = blockIdx.x * BM;
  const int bn   = blockIdx.y * BN;

  // Per-thread staging coordinates (k0-invariant parts)
  // A: 4 vec4 per thread over a 128x32 tile
  int a_r[4], a_c[4]; bool a_rok[4];
  // B: 4 vec4 per thread over a 32x128 tile
  int b_r[4], b_c[4]; bool b_cok[4];
  #pragma unroll
  for (int j = 0; j < 4; ++j) {
    int v = tid + j * 256;
    a_r[j] = v >> 3;            a_c[j] = (v & 7) << 2;
    a_rok[j] = (bm + a_r[j]) < M;
    b_r[j] = v >> 5;            b_c[j] = (v & 31) << 2;
    b_cok[j] = (bn + b_c[j]) < N;
  }

  float4 ra[4], rb[4];

  // Branch-free tile load: clamp addresses, zero-select out-of-range.
  auto load_tile = [&](int k0) {
    #pragma unroll
    for (int j = 0; j < 4; ++j) {
      int gr = imin(bm + a_r[j], M - 1);
      int gc = imin(k0 + a_c[j], K - 4);
      ra[j] = *(const float4*)(A + (size_t)gr * lda + gc);
    }
    #pragma unroll
    for (int j = 0; j < 4; ++j) {
      int gr = imin(k0 + b_r[j], K - 1);
      int gc = imin(bn + b_c[j], N - 4);
      rb[j] = *(const float4*)(B + (size_t)gr * ldb + gc);
    }
    #pragma unroll
    for (int j = 0; j < 4; ++j) {
      if (!(a_rok[j] && (k0 + a_c[j]) < K)) ra[j] = make_float4(0.f, 0.f, 0.f, 0.f);
      if (!(b_cok[j] && (k0 + b_r[j]) < K)) rb[j] = make_float4(0.f, 0.f, 0.f, 0.f);
    }
  };

  auto store_tile = [&]() {
    #pragma unroll
    for (int j = 0; j < 4; ++j) {
      v4h h = { (_Float16)ra[j].x, (_Float16)ra[j].y, (_Float16)ra[j].z, (_Float16)ra[j].w };
      *(v4h*)&sA[a_r[j]][a_c[j]] = h;
    }
    #pragma unroll
    for (int j = 0; j < 4; ++j) {
      sBt[b_c[j] + 0][b_r[j]] = (_Float16)rb[j].x;
      sBt[b_c[j] + 1][b_r[j]] = (_Float16)rb[j].y;
      sBt[b_c[j] + 2][b_r[j]] = (_Float16)rb[j].z;
      sBt[b_c[j] + 3][b_r[j]] = (_Float16)rb[j].w;
    }
  };

  v8f acc[2][4] = {};
  const int half = (lane < 16) ? 0 : 8;

  load_tile(0);
  int k0 = 0;
  while (true) {
    store_tile();
    __syncthreads();

    bool more = (k0 + BK) < K;
    if (more) load_tile(k0 + BK);   // global loads in flight during WMMA phase

    v16h afrag[2], bfrag[4];
    #pragma unroll
    for (int f = 0; f < 2; ++f)
      afrag[f] = frag_from(&sA[wm * 32 + f * 16 + (lane & 15)][0], half);
    #pragma unroll
    for (int f = 0; f < 4; ++f)
      bfrag[f] = frag_from(&sBt[wn * 64 + f * 16 + (lane & 15)][0], half);

    #pragma unroll
    for (int fm = 0; fm < 2; ++fm)
      #pragma unroll
      for (int fn = 0; fn < 4; ++fn)
        acc[fm][fn] = __builtin_amdgcn_wmma_f32_16x16x32_f16(
            false, afrag[fm], false, bfrag[fn], (short)0, acc[fm][fn], false, false);
    __syncthreads();

    if (!more) break;
    k0 += BK;
  }

  // Epilogue per C/D layout: VGPR r -> M = r (+8 for lanes 16..31), N = lane&15
  #pragma unroll
  for (int fm = 0; fm < 2; ++fm)
    #pragma unroll
    for (int fn = 0; fn < 4; ++fn) {
      int col = bn + wn * 64 + fn * 16 + (lane & 15);
      if (col >= N) continue;
      float bv = bias ? bias[col] : 0.f;
      int rbase = bm + wm * 32 + fm * 16 + ((lane >> 4) << 3);
      #pragma unroll
      for (int r = 0; r < 8; ++r) {
        int row = rbase + r;
        if (row >= M) continue;
        float v = acc[fm][fn][r] + bv;
        if (do_acc) v += C[(size_t)row * ldc + col];
        if (do_relu) v = fmaxf(v, 0.f);
        C[(size_t)row * ldc + col] = v;
      }
    }
}

// ---------------------------------------------------------------------------
// Edge phase: one wave32 per edge. r = h_src[src] - pred[dst]; LayerNorm over
// 256 via wave reduction; atomically accumulate into aggr[dst] and cnt[dst].
// ---------------------------------------------------------------------------
__global__ __launch_bounds__(256) void edge_ln_scatter(
    const int* __restrict__ src_idx, const int* __restrict__ dst_idx,
    const float* __restrict__ h_src, const float* __restrict__ pred,
    const float* __restrict__ ln_g, const float* __restrict__ ln_b,
    float* __restrict__ aggr, float* __restrict__ cnt, int E)
{
  int e    = (blockIdx.x * blockDim.x + threadIdx.x) >> 5;
  int lane = threadIdx.x & 31;
  if (e >= E) return;
  int s = src_idx[e];
  int d = dst_idx[e];

  const float4* xs = (const float4*)(h_src + (size_t)s * HID);
  const float4* pp = (const float4*)(pred  + (size_t)d * HID);
  float4 a0 = xs[lane * 2], a1 = xs[lane * 2 + 1];
  float4 p0 = pp[lane * 2], p1 = pp[lane * 2 + 1];

  float r[8] = { a0.x - p0.x, a0.y - p0.y, a0.z - p0.z, a0.w - p0.w,
                 a1.x - p1.x, a1.y - p1.y, a1.z - p1.z, a1.w - p1.w };
  float s1 = 0.f, s2 = 0.f;
  #pragma unroll
  for (int i = 0; i < 8; ++i) { s1 += r[i]; s2 += r[i] * r[i]; }
  #pragma unroll
  for (int off = 16; off > 0; off >>= 1) {
    s1 += __shfl_xor(s1, off, 32);
    s2 += __shfl_xor(s2, off, 32);
  }
  float mu  = s1 * (1.f / HID);
  float var = s2 * (1.f / HID) - mu * mu;
  float inv = rsqrtf(var + LN_EPS);

  float* ag = aggr + (size_t)d * HID + lane * 8;
  const float* g = ln_g + lane * 8;
  const float* b = ln_b + lane * 8;
  #pragma unroll
  for (int i = 0; i < 8; ++i)
    atomicAdd(&ag[i], (r[i] - mu) * inv * g[i] + b[i]);
  if (lane == 0) atomicAdd(&cnt[d], 1.0f);
}

// aggr[n][c] /= max(cnt[n], 1)
__global__ void finalize_mean(float* __restrict__ aggr, const float* __restrict__ cnt, int total)
{
  int i = blockIdx.x * blockDim.x + threadIdx.x;
  if (i < total) aggr[i] /= fmaxf(cnt[i >> 8], 1.0f);
}

// h[row] = relu(LN(new_h[row]; g,b)) + h[row]   (one wave32 per row)
__global__ __launch_bounds__(256) void ln_relu_residual(
    const float* __restrict__ newh, const float* __restrict__ g,
    const float* __restrict__ b, float* __restrict__ h, int rows)
{
  int row  = (blockIdx.x * blockDim.x + threadIdx.x) >> 5;
  int lane = threadIdx.x & 31;
  if (row >= rows) return;
  const float* x = newh + (size_t)row * HID + lane * 8;
  float r[8]; float s1 = 0.f, s2 = 0.f;
  #pragma unroll
  for (int i = 0; i < 8; ++i) { r[i] = x[i]; s1 += r[i]; s2 += r[i] * r[i]; }
  #pragma unroll
  for (int off = 16; off > 0; off >>= 1) {
    s1 += __shfl_xor(s1, off, 32);
    s2 += __shfl_xor(s2, off, 32);
  }
  float mu  = s1 * (1.f / HID);
  float var = s2 * (1.f / HID) - mu * mu;
  float inv = rsqrtf(var + LN_EPS);
  float* hp = h + (size_t)row * HID + lane * 8;
  #pragma unroll
  for (int i = 0; i < 8; ++i) {
    float v = (r[i] - mu) * inv * g[lane * 8 + i] + b[lane * 8 + i];
    hp[i] = fmaxf(v, 0.f) + hp[i];
  }
}

// out[row] = z[row,:128] . w2 + b2   (one wave32 per row, 4 elems/lane)
__global__ __launch_bounds__(256) void head_dot128(
    const float* __restrict__ z, const float* __restrict__ w2,
    const float* __restrict__ b2, float* __restrict__ out, int rows)
{
  int row  = (blockIdx.x * blockDim.x + threadIdx.x) >> 5;
  int lane = threadIdx.x & 31;
  if (row >= rows) return;
  const float* zp = z + (size_t)row * 128 + lane * 4;
  float s = zp[0] * w2[lane * 4 + 0] + zp[1] * w2[lane * 4 + 1] +
            zp[2] * w2[lane * 4 + 2] + zp[3] * w2[lane * 4 + 3];
  #pragma unroll
  for (int off = 16; off > 0; off >>= 1) s += __shfl_xor(s, off, 32);
  if (lane == 0) out[row] = s + b2[0];
}

// ---------------------------------------------------------------------------
// Host side
// ---------------------------------------------------------------------------
static void gemm(hipStream_t st, const float* A, int lda, const float* B, int ldb,
                 const float* bias, float* C, int ldc, int M, int N, int K,
                 int relu, int acc)
{
  dim3 g((M + BM - 1) / BM, (N + BN - 1) / BN);
  gemm_wmma_f16<<<g, 256, 0, st>>>(A, lda, B, ldb, bias, C, ldc, M, N, K, relu, acc);
}

// One PRMP conv: predicted MLP deduped to per-dst-node, per-edge LN scatter,
// mean, and update GEMM (concat done as two accumulated GEMMs).
static void conv(hipStream_t st,
                 const float* h_src, const float* h_dst, int n_dst,
                 const int* src_idx, const int* dst_idx, int E,
                 const float* p1W, const float* p1b,
                 const float* p2W, const float* p2b,
                 const float* lng, const float* lnb,
                 const float* updW, const float* updb,
                 float* hidden, float* pred, float* aggr, float* cnt,
                 float* out)
{
  // predicted = relu(h_dst@W1+b1)@W2+b2, computed once per destination node
  gemm(st, h_dst, HID, p1W, HID, p1b, hidden, HID, n_dst, HID, HID, 1, 0);
  gemm(st, hidden, HID, p2W, HID, p2b, pred,   HID, n_dst, HID, HID, 0, 0);

  hipMemsetAsync(aggr, 0, (size_t)n_dst * HID * sizeof(float), st);
  hipMemsetAsync(cnt,  0, (size_t)n_dst * sizeof(float), st);

  edge_ln_scatter<<<(E + 7) / 8, 256, 0, st>>>(src_idx, dst_idx, h_src, pred,
                                               lng, lnb, aggr, cnt, E);
  int total = n_dst * HID;
  finalize_mean<<<(total + 255) / 256, 256, 0, st>>>(aggr, cnt, total);

  // out = [h_dst, aggr] @ updW + updb  == h_dst@updW[:256] + aggr@updW[256:] + b
  gemm(st, h_dst, HID, updW,             HID, updb,    out, HID, n_dst, HID, HID, 0, 0);
  gemm(st, aggr,  HID, updW + HID * HID, HID, nullptr, out, HID, n_dst, HID, HID, 0, 1);
}

extern "C" void kernel_launch(void* const* d_in, const int* in_sizes, int n_in,
                              void* d_out, int out_size, void* d_ws, size_t ws_size,
                              hipStream_t stream)
{
  const float* x_drv  = (const float*)d_in[0];
  const float* x_rc   = (const float*)d_in[1];
  const int*   dr_src = (const int*)d_in[2];
  const int*   dr_dst = (const int*)d_in[3];
  const int*   rd_src = (const int*)d_in[4];
  const int*   rd_dst = (const int*)d_in[5];

  const int NDRV  = in_sizes[0] / 32;   // 50000
  const int NRACE = in_sizes[1] / 16;   // 10000
  const int E     = in_sizes[2];        // 200000

  // Params: jax pytree flatten order (dict keys sorted): enc -> head -> layers.
  // enc: drivers(W,b), races(W,b); head: l1(W,b), l2(W,b);
  // layer: conv{dr: ln_b,ln_g,p1(W,b),p2(W,b),upd(W,b); rd: same}, ln{drv(g,b), rc(g,b)}
  #define P(i) ((const float*)d_in[6 + (i)])
  const float* encDrvW = P(0); const float* encDrvB = P(1);
  const float* encRcW  = P(2); const float* encRcB  = P(3);
  const float* hl1W    = P(4); const float* hl1B    = P(5);
  const float* hl2W    = P(6); const float* hl2B    = P(7);
  #define LB(l) (8 + 20 * (l))

  // Workspace carve (floats)
  float* w = (float*)d_ws;
  size_t o = 0;
  float* h_drv   = w + o; o += (size_t)NDRV  * HID;
  float* h_rc    = w + o; o += (size_t)NRACE * HID;
  float* new_drv = w + o; o += (size_t)NDRV  * HID;
  float* new_rc  = w + o; o += (size_t)NRACE * HID;
  float* hidden  = w + o; o += (size_t)NDRV  * HID;  // also reused as head z
  float* pred    = w + o; o += (size_t)NDRV  * HID;
  float* aggr    = w + o; o += (size_t)NDRV  * HID;
  float* cnt     = w + o; o += (size_t)((NDRV + 63) & ~63);
  (void)ws_size; (void)n_in;

  // Encoders: h = relu(x @ We + be)
  gemm(stream, x_drv, 32, encDrvW, HID, encDrvB, h_drv, HID, NDRV,  HID, 32, 1, 0);
  gemm(stream, x_rc,  16, encRcW,  HID, encRcB,  h_rc,  HID, NRACE, HID, 16, 1, 0);

  for (int l = 0; l < 2; ++l) {
    const int B0 = LB(l);
    // drivers -> races (dst = races)
    conv(stream, h_drv, h_rc, NRACE, dr_src, dr_dst, E,
         P(B0 + 2), P(B0 + 3), P(B0 + 4), P(B0 + 5), P(B0 + 1), P(B0 + 0),
         P(B0 + 6), P(B0 + 7), hidden, pred, aggr, cnt, new_rc);
    // races -> drivers (dst = drivers)
    conv(stream, h_rc, h_drv, NDRV, rd_src, rd_dst, E,
         P(B0 + 10), P(B0 + 11), P(B0 + 12), P(B0 + 13), P(B0 + 9), P(B0 + 8),
         P(B0 + 14), P(B0 + 15), hidden, pred, aggr, cnt, new_drv);
    // h = relu(LN(new_h)) + h   (both node types read old h above; update after)
    ln_relu_residual<<<(NDRV  + 7) / 8, 256, 0, stream>>>(new_drv, P(B0 + 16), P(B0 + 17), h_drv, NDRV);
    ln_relu_residual<<<(NRACE + 7) / 8, 256, 0, stream>>>(new_rc,  P(B0 + 18), P(B0 + 19), h_rc,  NRACE);
  }

  // Head: z = relu(h_drv @ W1 + b1) [NDRV,128]; out = z @ W2 + b2
  float* z = hidden;
  gemm(stream, h_drv, HID, hl1W, 128, hl1B, z, 128, NDRV, 128, HID, 1, 0);
  head_dot128<<<(NDRV + 7) / 8, 256, 0, stream>>>(z, hl2W, hl2B, (float*)d_out, NDRV);
}
